// DotProductAttention_37915971289585
// MI455X (gfx1250) — matrix operand order repaired and checked by reading, not compiled, not run on hardware
//
#include <hip/hip_runtime.h>
#include <hip/hip_bf16.h>

// ---------------------------------------------------------------------------
// Transformer encoder block forward (MI455X / gfx1250, wave32, WMMA f16,
// async global->LDS staging).  N=2, T=2048, H=16, Dh=64, D=1024, DFF=4096
// ---------------------------------------------------------------------------

typedef __attribute__((ext_vector_type(16))) _Float16 v16h;
typedef __attribute__((ext_vector_type(8)))  _Float16 v8h;
typedef __attribute__((ext_vector_type(8)))  float    v8f;

#define TT      2048
#define NB      2
#define NH      16
#define DH      64
#define DM      1024
#define DFF_    4096
#define MTOT    (NB * TT)          // 4096 token rows

__device__ __forceinline__ v8f wmma16(v16h a, v16h b, v8f c) {
    return __builtin_amdgcn_wmma_f32_16x16x32_f16(
        /*neg_a=*/false, a, /*neg_b=*/false, b,
        /*c_mod=*/(short)0, c, /*reuse_a=*/false, /*reuse_b=*/false);
}

// Async copy of 16 bytes global -> LDS (ASYNCcnt-tracked, gfx1250).
// LDS operand = low 32 bits of the generic pointer.
__device__ __forceinline__ void async_b128(const void* gptr, void* lptr) {
    unsigned lds = (unsigned)(size_t)lptr;
    asm volatile("global_load_async_to_lds_b128 %0, %1, off"
                 :: "v"(lds), "v"(gptr) : "memory");
}
__device__ __forceinline__ void wait_async0() {
    asm volatile("s_wait_asynccnt 0" ::: "memory");
}

// A fragment (16x32 f16): lane = row M (lane&15), g = lane>>4 selects K-half.
// K map per ISA: VGPR v<4 -> K=2v, v>=4 -> K=16+2(v-4), +8g, half h -> +h.
__device__ __forceinline__ v16h frag_A(const _Float16* line, int g) {
    v16h f;
#pragma unroll
    for (int v = 0; v < 8; ++v) {
        int k = ((v < 4) ? (2 * v) : (16 + 2 * (v - 4))) + 8 * g;
        f[2 * v]     = line[k];
        f[2 * v + 1] = line[k + 1];
    }
    return f;
}

// B fragment (32x16 f16): lane = col N (lane&15); K map same as frag_A.
__device__ __forceinline__ v16h frag_B(const _Float16* line, int g) {
    v16h f;
    const _Float16* p = line + 8 * g;
#pragma unroll
    for (int v = 0; v < 8; ++v) {
        int k = (v < 4) ? (2 * v) : (16 + 2 * (v - 4));
        f[2 * v]     = p[k];
        f[2 * v + 1] = p[k + 1];
    }
    return f;
}

__device__ __forceinline__ v8f zero8() {
    v8f z = {0.f, 0.f, 0.f, 0.f, 0.f, 0.f, 0.f, 0.f};
    return z;
}

// ---------------------------------------------------------------------------
// Shared GEMM mainloop: C(128x64 block tile) = A[M,K](f16,row) * B[K,N](f16,row)
// 8 waves in 4x2; wave (wm=w>>1, wn=w&1) computes a 32x32 C sub-tile
// (2 A-frags x 2 B-frags -> 4 WMMAs per 32-deep K chunk).
// A tile staged with async global->LDS; B tile staged transposed.
// ---------------------------------------------------------------------------
__device__ __forceinline__ void gemm_mainloop(
    const _Float16* __restrict__ A, const _Float16* __restrict__ B,
    int N, int K, int m0, int n0,
    _Float16* As /*128x40*/, _Float16* Bs /*64x40, transposed [n][k]*/,
    v8f& acc00, v8f& acc01, v8f& acc10, v8f& acc11)
{
    const int tid  = threadIdx.x;
    const int lane = tid & 31;
    const int wave = tid >> 5;
    const int wm   = wave >> 1;      // 0..3 -> 32-row strip
    const int wn   = wave & 1;       // 0..1 -> 32-col strip
    const int g    = lane >> 4;
    const int l15  = lane & 15;

    acc00 = zero8(); acc01 = zero8(); acc10 = zero8(); acc11 = zero8();

    for (int kk = 0; kk < K; kk += 32) {
        __syncthreads();
        // stage A tile 128x32 via async copy (512 x 16B segments, 2/thread)
#pragma unroll
        for (int s = 0; s < 2; ++s) {
            int seg = tid + s * 256;
            int m   = seg >> 2;
            int ks  = (seg & 3) * 8;
            async_b128(A + (size_t)(m0 + m) * K + kk + ks, As + m * 40 + ks);
        }
        // stage B tile 32x64, transposed into Bs[n][k]
        {
            int kr = tid >> 3;
            int ns = (tid & 7) * 8;
            v8h bv = *(const v8h*)(B + (size_t)(kk + kr) * N + n0 + ns);
#pragma unroll
            for (int j = 0; j < 8; ++j) Bs[(ns + j) * 40 + kr] = bv[j];
        }
        wait_async0();
        __syncthreads();

        v16h a0 = frag_A(As + (wm * 32 + l15) * 40, g);
        v16h a1 = frag_A(As + (wm * 32 + 16 + l15) * 40, g);
        v16h b0 = frag_B(Bs + (wn * 32 + l15) * 40, g);
        v16h b1 = frag_B(Bs + (wn * 32 + 16 + l15) * 40, g);
        acc00 = wmma16(a0, b0, acc00);
        acc01 = wmma16(a0, b1, acc01);
        acc10 = wmma16(a1, b0, acc10);
        acc11 = wmma16(a1, b1, acc11);
    }
}

// ---------------------------------------------------------------------------
// Kernel: f32 -> f16 conversion (grid-stride)
// ---------------------------------------------------------------------------
__global__ void cvt_f32_f16(const float* __restrict__ in,
                            _Float16* __restrict__ out, int n) {
    for (int i = blockIdx.x * blockDim.x + threadIdx.x; i < n;
         i += gridDim.x * blockDim.x)
        out[i] = (_Float16)in[i];
}

// ---------------------------------------------------------------------------
// Kernel: QKV projection.  C[4096,3072] = Xh @ Wqkv + b, scattered into
// Q/K/V[N,H,T,Dh] f16.  Column c = h*192 + d*3 + s  (s: 0=q,1=k,2=v).
// ---------------------------------------------------------------------------
__global__ void __launch_bounds__(256)
gemm_qkv_kernel(const _Float16* __restrict__ A, const _Float16* __restrict__ B,
                const float* __restrict__ bias,
                _Float16* __restrict__ Qh, _Float16* __restrict__ Kh,
                _Float16* __restrict__ Vh)
{
    __shared__ _Float16 As[128 * 40];
    __shared__ _Float16 Bs[64 * 40];
    const int m0 = blockIdx.y * 128, n0 = blockIdx.x * 64;
    v8f a00, a01, a10, a11;
    gemm_mainloop(A, B, 3 * DM, DM, m0, n0, As, Bs, a00, a01, a10, a11);

    const int lane = threadIdx.x & 31, wave = threadIdx.x >> 5;
    const int wm = wave >> 1, wn = wave & 1, g = lane >> 4, l15 = lane & 15;
#pragma unroll
    for (int sm = 0; sm < 2; ++sm) {
#pragma unroll
        for (int sn = 0; sn < 2; ++sn) {
            v8f acc = sm ? (sn ? a11 : a10) : (sn ? a01 : a00);
#pragma unroll
            for (int r = 0; r < 8; ++r) {
                int row = m0 + wm * 32 + sm * 16 + r + 8 * g;
                int col = n0 + wn * 32 + sn * 16 + l15;
                float val = acc[r] + bias[col];
                int hh  = col / 192;
                int rem = col - hh * 192;
                int d   = rem / 3;
                int s   = rem - d * 3;
                int nn  = row >> 11;           // / T
                int t   = row & (TT - 1);
                size_t di = (((size_t)nn * NH + hh) * TT + t) * DH + d;
                _Float16 hv = (_Float16)val;
                if (s == 0)      Qh[di] = hv;
                else if (s == 1) Kh[di] = hv;
                else             Vh[di] = hv;
            }
        }
    }
}

// ---------------------------------------------------------------------------
// Kernel: generic FF GEMM with bias (+optional relu), f16 and/or f32 output.
// ---------------------------------------------------------------------------
__global__ void __launch_bounds__(256)
gemm_ff_kernel(const _Float16* __restrict__ A, const _Float16* __restrict__ B,
               const float* __restrict__ bias,
               _Float16* __restrict__ outH, float* __restrict__ outF,
               int N, int K, int relu)
{
    __shared__ _Float16 As[128 * 40];
    __shared__ _Float16 Bs[64 * 40];
    const int m0 = blockIdx.y * 128, n0 = blockIdx.x * 64;
    v8f a00, a01, a10, a11;
    gemm_mainloop(A, B, N, K, m0, n0, As, Bs, a00, a01, a10, a11);

    const int lane = threadIdx.x & 31, wave = threadIdx.x >> 5;
    const int wm = wave >> 1, wn = wave & 1, g = lane >> 4, l15 = lane & 15;
#pragma unroll
    for (int sm = 0; sm < 2; ++sm) {
#pragma unroll
        for (int sn = 0; sn < 2; ++sn) {
            v8f acc = sm ? (sn ? a11 : a10) : (sn ? a01 : a00);
#pragma unroll
            for (int r = 0; r < 8; ++r) {
                int row = m0 + wm * 32 + sm * 16 + r + 8 * g;
                int col = n0 + wn * 32 + sn * 16 + l15;
                float val = acc[r] + bias[col];
                if (relu) val = fmaxf(val, 0.f);
                size_t idx = (size_t)row * N + col;
                if (outH) outH[idx] = (_Float16)val;
                if (outF) outF[idx] = val;
            }
        }
    }
}

// ---------------------------------------------------------------------------
// Kernel: flash attention.  grid=(T/128, N*H), block=256 (8 waves).
// Each wave owns a 16-row q tile; block stages 32-key K/V tiles in LDS
// (K via async copy, V transposed manually).
// ---------------------------------------------------------------------------
__global__ void __launch_bounds__(256)
attn_kernel(const _Float16* __restrict__ Qh, const _Float16* __restrict__ Kh,
            const _Float16* __restrict__ Vh, const float* __restrict__ mask,
            float* __restrict__ ctx)
{
    __shared__ _Float16 Ks[32 * 72];        // K tile, row-major [key][d]
    __shared__ _Float16 Vts[64 * 40];       // V tile transposed [d][key]
    __shared__ _Float16 Ps[8 * 16 * 40];    // per-wave P scratch [m][key]

    const int tid  = threadIdx.x;
    const int lane = tid & 31;
    const int wave = tid >> 5;
    const int g    = lane >> 4;
    const int l15  = lane & 15;
    const int nh   = blockIdx.y;
    const int n    = nh >> 4;
    const int h    = nh & 15;
    const int qtile = blockIdx.x * 128 + wave * 16;

    const _Float16* Qb = Qh + ((size_t)nh * TT + qtile) * DH;
    const _Float16* Kb = Kh + (size_t)nh * TT * DH;
    const _Float16* Vb = Vh + (size_t)nh * TT * DH;
    _Float16* myPs = Ps + wave * 16 * 40;

    // Q fragments, loaded once (rows qtile..+15, d split in two K=32 chunks)
    v16h aQ0 = frag_A(Qb + (size_t)l15 * DH, g);
    v16h aQ1 = frag_A(Qb + (size_t)l15 * DH + 32, g);

    float mstate[8], lstate[8], mbias[8];
#pragma unroll
    for (int r = 0; r < 8; ++r) {
        int row = qtile + r + 8 * g;
        mbias[r]  = (1.0f - mask[n * TT + row]) * -1e9f;
        mstate[r] = -1e30f;
        lstate[r] = 0.0f;
    }
    v8f o0 = zero8(), o1 = zero8(), o2 = zero8(), o3 = zero8();
    const float scale = 0.125f;   // 1/sqrt(64)

    for (int kb = 0; kb < TT / 32; ++kb) {
        __syncthreads();
        {   // stage K (async, row-major) and V (transposed)
            int kr = tid >> 3;
            int ds = (tid & 7) * 8;
            async_b128(Kb + (size_t)(kb * 32 + kr) * DH + ds,
                       Ks + kr * 72 + ds);
            v8h vv = *(const v8h*)(Vb + (size_t)(kb * 32 + kr) * DH + ds);
#pragma unroll
            for (int j = 0; j < 8; ++j) Vts[(ds + j) * 40 + kr] = vv[j];
        }
        wait_async0();
        __syncthreads();

        // S = Q @ K^T : two key sub-tiles, K-dim 64 in two chunks
        v8f s0 = zero8(), s1 = zero8();
        s0 = wmma16(aQ0, frag_B(Ks + l15 * 72, g), s0);
        s0 = wmma16(aQ1, frag_B(Ks + l15 * 72 + 32, g), s0);
        s1 = wmma16(aQ0, frag_B(Ks + (16 + l15) * 72, g), s1);
        s1 = wmma16(aQ1, frag_B(Ks + (16 + l15) * 72 + 32, g), s1);

        // online softmax (rows r+8g, cols spread over 16-lane groups)
#pragma unroll
        for (int r = 0; r < 8; ++r) {
            float v0 = s0[r] * scale + mbias[r];
            float v1 = s1[r] * scale + mbias[r];
            float mx = fmaxf(v0, v1);
#pragma unroll
            for (int off = 1; off < 16; off <<= 1)
                mx = fmaxf(mx, __shfl_xor(mx, off, 32));
            float mnew = fmaxf(mstate[r], mx);
            float p0 = __expf(v0 - mnew);
            float p1 = __expf(v1 - mnew);
            float sum = p0 + p1;
#pragma unroll
            for (int off = 1; off < 16; off <<= 1)
                sum += __shfl_xor(sum, off, 32);
            float alpha = __expf(mstate[r] - mnew);
            lstate[r] = lstate[r] * alpha + sum;
            mstate[r] = mnew;
            o0[r] *= alpha; o1[r] *= alpha; o2[r] *= alpha; o3[r] *= alpha;
            myPs[(r + 8 * g) * 40 + l15]      = (_Float16)p0;
            myPs[(r + 8 * g) * 40 + 16 + l15] = (_Float16)p1;
        }
        __syncthreads();   // make P visible across lanes before re-fragment

        // O += P(16x32) @ V(32x64)
        v16h aP = frag_A(myPs + l15 * 40, g);
        o0 = wmma16(aP, frag_B(Vts + (0 * 16 + l15) * 40, g), o0);
        o1 = wmma16(aP, frag_B(Vts + (1 * 16 + l15) * 40, g), o1);
        o2 = wmma16(aP, frag_B(Vts + (2 * 16 + l15) * 40, g), o2);
        o3 = wmma16(aP, frag_B(Vts + (3 * 16 + l15) * 40, g), o3);
    }

    // epilogue: ctx[n, t, h*64 + d] = O / l
#pragma unroll
    for (int r = 0; r < 8; ++r) {
        float inv = 1.0f / lstate[r];
        int row = qtile + r + 8 * g;
        float* dst = ctx + ((size_t)n * TT + row) * DM + h * DH;
        dst[0 * 16 + l15] = o0[r] * inv;
        dst[1 * 16 + l15] = o1[r] * inv;
        dst[2 * 16 + l15] = o2[r] * inv;
        dst[3 * 16 + l15] = o3[r] * inv;
    }
}

// ---------------------------------------------------------------------------
// Kernel: fused residual-add + LayerNorm (row of 1024).  One block per row.
// ---------------------------------------------------------------------------
__global__ void __launch_bounds__(256)
ln_kernel(const float* __restrict__ a, const float* __restrict__ badd,
          const float* __restrict__ gamma, const float* __restrict__ beta,
          float* __restrict__ outF, _Float16* __restrict__ outH)
{
    __shared__ float red[256];
    const int row = blockIdx.x;
    const int tid = threadIdx.x;
    const float* pa = a + (size_t)row * DM;
    const float* pb = badd + (size_t)row * DM;

    float v[4];
    float s = 0.f;
#pragma unroll
    for (int i = 0; i < 4; ++i) {
        int c = tid + 256 * i;
        v[i] = pa[c] + pb[c];
        s += v[i];
    }
    red[tid] = s;
    __syncthreads();
    for (int o = 128; o > 0; o >>= 1) {
        if (tid < o) red[tid] += red[tid + o];
        __syncthreads();
    }
    float mu = red[0] * (1.0f / DM);
    __syncthreads();

    float s2 = 0.f;
#pragma unroll
    for (int i = 0; i < 4; ++i) {
        float d = v[i] - mu;
        s2 += d * d;
    }
    red[tid] = s2;
    __syncthreads();
    for (int o = 128; o > 0; o >>= 1) {
        if (tid < o) red[tid] += red[tid + o];
        __syncthreads();
    }
    float var = red[0] * (1.0f / DM);
    float rs  = rsqrtf(var + 1e-3f);

#pragma unroll
    for (int i = 0; i < 4; ++i) {
        int c = tid + 256 * i;
        float y = (v[i] - mu) * rs * gamma[c] + beta[c];
        if (outF) outF[(size_t)row * DM + c] = y;
        if (outH) outH[(size_t)row * DM + c] = (_Float16)y;
    }
}

// ---------------------------------------------------------------------------
// Host orchestration
// ---------------------------------------------------------------------------
extern "C" void kernel_launch(void* const* d_in, const int* in_sizes, int n_in,
                              void* d_out, int out_size, void* d_ws,
                              size_t ws_size, hipStream_t stream) {
    (void)in_sizes; (void)n_in; (void)out_size; (void)ws_size;

    const float* x     = (const float*)d_in[0];
    const float* mask  = (const float*)d_in[1];
    const float* w_qkv = (const float*)d_in[2];
    const float* b_qkv = (const float*)d_in[3];
    const float* w_ff  = (const float*)d_in[4];
    const float* b_ff  = (const float*)d_in[5];
    const float* w_out = (const float*)d_in[6];
    const float* b_out = (const float*)d_in[7];
    const float* ln1_g = (const float*)d_in[8];
    const float* ln1_b = (const float*)d_in[9];
    const float* ln2_g = (const float*)d_in[10];
    const float* ln2_b = (const float*)d_in[11];
    float* out = (float*)d_out;

    char* ws = (char*)d_ws;
    size_t off = 0;
    auto alloc = [&](size_t bytes) -> void* {
        void* p = ws + off;
        off += (bytes + 255) & ~(size_t)255;
        return p;
    };

    _Float16* x_h    = (_Float16*)alloc((size_t)MTOT * DM * 2);   // reused as h_h
    _Float16* wqkv_h = (_Float16*)alloc((size_t)DM * 3 * DM * 2);
    _Float16* wff_h  = (_Float16*)alloc((size_t)DM * DFF_ * 2);
    _Float16* wout_h = (_Float16*)alloc((size_t)DFF_ * DM * 2);
    _Float16* Qh     = (_Float16*)alloc((size_t)MTOT * DM * 2);
    _Float16* Kh     = (_Float16*)alloc((size_t)MTOT * DM * 2);
    _Float16* Vh     = (_Float16*)alloc((size_t)MTOT * DM * 2);
    float*    ctx    = (float*)alloc((size_t)MTOT * DM * 4);      // reused as fo
    float*    h_f32  = (float*)alloc((size_t)MTOT * DM * 4);
    _Float16* ff_h   = (_Float16*)alloc((size_t)MTOT * DFF_ * 2);

    _Float16* h_h = x_h;   // alias: x_h dead after QKV GEMM
    float*    fo  = ctx;   // alias: ctx dead after LN1

    // 1. convert operands to f16
    cvt_f32_f16<<<4096, 256, 0, stream>>>(x, x_h, MTOT * DM);
    cvt_f32_f16<<<4096, 256, 0, stream>>>(w_qkv, wqkv_h, DM * 3 * DM);
    cvt_f32_f16<<<4096, 256, 0, stream>>>(w_ff, wff_h, DM * DFF_);
    cvt_f32_f16<<<4096, 256, 0, stream>>>(w_out, wout_h, DFF_ * DM);

    // 2. QKV projection -> Q/K/V [N,H,T,Dh] f16
    gemm_qkv_kernel<<<dim3(3 * DM / 64, MTOT / 128), 256, 0, stream>>>(
        x_h, wqkv_h, b_qkv, Qh, Kh, Vh);

    // 3. flash attention -> ctx [N,T,D] f32
    attn_kernel<<<dim3(TT / 128, NB * NH), 256, 0, stream>>>(
        Qh, Kh, Vh, mask, ctx);

    // 4. LN1(x + ctx) -> h_f32, h_h
    ln_kernel<<<MTOT, 256, 0, stream>>>(x, ctx, ln1_g, ln1_b, h_f32, h_h);

    // 5. relu(h @ w_ff + b_ff) -> ff_h (f16)
    gemm_ff_kernel<<<dim3(DFF_ / 64, MTOT / 128), 256, 0, stream>>>(
        h_h, wff_h, b_ff, ff_h, (float*)nullptr, DFF_, DM, 1);

    // 6. ff @ w_out + b_out -> fo (f32)
    gemm_ff_kernel<<<dim3(DM / 64, MTOT / 128), 256, 0, stream>>>(
        ff_h, wout_h, b_out, (_Float16*)nullptr, fo, DM, DFF_, 0);

    // 7. LN2(h + fo) -> out
    ln_kernel<<<MTOT, 256, 0, stream>>>(h_f32, fo, ln2_g, ln2_b, out,
                                        (_Float16*)nullptr);
}
